// RNNSequenceEmbedder_41455024341192
// MI455X (gfx1250) — compile-verified, matrix operand order checked by reading, not run on hardware
//
#include <hip/hip_runtime.h>

typedef __attribute__((ext_vector_type(8)))  float  v8f;
typedef __attribute__((ext_vector_type(16))) __bf16 v16bf;
typedef __attribute__((ext_vector_type(8)))  __bf16 v8bf;

static constexpr int Bq = 64;     // batch
static constexpr int Tq = 256;    // time steps
static constexpr int Dq = 512;    // embed dim
static constexpr int Hq = 1024;   // hidden
static constexpr int G  = 4096;   // 4*H gates

// dynamic LDS layout for the step kernel
static constexpr int HBUF_BYTES = Bq * Hq * 2;           // 131072 (staged h, bf16)
static constexpr int WBUF_BYTES = 64 * Hq * 2;           // 131072 (staged 64 Whh rows)
static constexpr int GBUF_STRIDE = 68;                   // padded to dodge bank conflicts
static constexpr int STEP_LDS = HBUF_BYTES + WBUF_BYTES + 64 * GBUF_STRIDE * 4; // 279552

__device__ __forceinline__ v16bf cat8(v8bf lo, v8bf hi) {
  return __builtin_shufflevector(lo, hi, 0,1,2,3,4,5,6,7,8,9,10,11,12,13,14,15);
}
__device__ __forceinline__ float sigm(float x) { return 1.0f / (1.0f + __expf(-x)); }

// async global -> LDS, 16 bytes per lane; tracked by ASYNCcnt, bypasses VGPRs
__device__ __forceinline__ void async_copy16(unsigned lds_addr, const void* gptr) {
  asm volatile("global_load_async_to_lds_b128 %0, %1, off"
               :: "v"(lds_addr), "v"((unsigned long long)(uintptr_t)gptr)
               : "memory");
}
__device__ __forceinline__ void wait_async0() {
  asm volatile("s_wait_asynccnt 0x0" ::: "memory");
}

// ------------- prologue: weights fp32->bf16, bias sum, zero h0/c0 -------------
__global__ void prep_kernel(const float* __restrict__ WihF, const float* __restrict__ WhhF,
                            const float* __restrict__ bih,  const float* __restrict__ bhh,
                            __bf16* __restrict__ Wih, __bf16* __restrict__ Whh,
                            float* __restrict__ bias, __bf16* __restrict__ h0,
                            float* __restrict__ c0) {
  size_t i = (size_t)blockIdx.x * blockDim.x + threadIdx.x;
  if (i < (size_t)G * Dq) Wih[i] = (__bf16)WihF[i];
  if (i < (size_t)G * Hq) Whh[i] = (__bf16)WhhF[i];
  if (i < (size_t)G)      bias[i] = bih[i] + bhh[i];
  if (i < (size_t)Bq * Hq) { h0[i] = (__bf16)0.0f; c0[i] = 0.0f; }
}

// ------------- embedding gather -> bf16, layout X[t][b][d] -------------
__global__ void embed_kernel(const int* __restrict__ tokens, const float* __restrict__ emb,
                             __bf16* __restrict__ X) {
  size_t gid = (size_t)blockIdx.x * blockDim.x + threadIdx.x;  // one 8-elem chunk
  int row = (int)(gid >> 6);            // 64 chunks of 8 per 512-wide row
  int d8  = (int)(gid & 63) << 3;
  int t = row >> 6, b = row & 63;       // row = t*64 + b
  int tok = tokens[b * Tq + t];
  const float4* s = reinterpret_cast<const float4*>(emb + (size_t)tok * Dq + d8);
  float4 lo = s[0], hi = s[1];
  v8bf o;
  o[0]=(__bf16)lo.x; o[1]=(__bf16)lo.y; o[2]=(__bf16)lo.z; o[3]=(__bf16)lo.w;
  o[4]=(__bf16)hi.x; o[5]=(__bf16)hi.y; o[6]=(__bf16)hi.z; o[7]=(__bf16)hi.w;
  *reinterpret_cast<v8bf*>(X + (size_t)row * Dq + d8) = o;
}

// ------------- big parallel GEMM: GX = X @ Wih^T + bias -------------
// 2x2 register tiles per wave (32x32 output): each fragment feeds 2 WMMAs.
// 8 waves/WG share one 32-row A strip (L0 reuse). 512x128 wave-blocks.
__global__ __launch_bounds__(256) void gemmx_kernel(const __bf16* __restrict__ X,
                                                    const __bf16* __restrict__ Wih,
                                                    const float* __restrict__ bias,
                                                    float* __restrict__ GX) {
  const int wave = threadIdx.x >> 5, lane = threadIdx.x & 31;
  const int blk = blockIdx.x * 8 + wave;
  const int colBlk = blk & 127;         // 4096/32
  const int rowBlk = blk >> 7;          // 16384/32
  const int rowBase = rowBlk << 5, colBase = colBlk << 5;
  const int lm = lane & 15, kh = (lane >> 4) << 3;   // ISA A-fragment layout
  const __bf16* ap0 = X   + (size_t)(rowBase + lm) * Dq + kh;
  const __bf16* ap1 = ap0 + (size_t)16 * Dq;
  const __bf16* bp0 = Wih + (size_t)(colBase + lm) * Dq + kh;
  const __bf16* bp1 = bp0 + (size_t)16 * Dq;
  v8f acc00 = {}, acc01 = {}, acc10 = {}, acc11 = {};
#pragma unroll 2
  for (int k = 0; k < Dq; k += 32) {
    v16bf a0 = cat8(*(const v8bf*)(ap0 + k), *(const v8bf*)(ap0 + k + 16));
    v16bf a1 = cat8(*(const v8bf*)(ap1 + k), *(const v8bf*)(ap1 + k + 16));
    v16bf b0 = cat8(*(const v8bf*)(bp0 + k), *(const v8bf*)(bp0 + k + 16));
    v16bf b1 = cat8(*(const v8bf*)(bp1 + k), *(const v8bf*)(bp1 + k + 16));
    acc00 = __builtin_amdgcn_wmma_f32_16x16x32_bf16(false, a0, false, b0, (short)0, acc00, false, false);
    acc01 = __builtin_amdgcn_wmma_f32_16x16x32_bf16(false, a0, false, b1, (short)0, acc01, false, false);
    acc10 = __builtin_amdgcn_wmma_f32_16x16x32_bf16(false, a1, false, b0, (short)0, acc10, false, false);
    acc11 = __builtin_amdgcn_wmma_f32_16x16x32_bf16(false, a1, false, b1, (short)0, acc11, false, false);
  }
  const float bv0 = bias[colBase + lm];
  const float bv1 = bias[colBase + 16 + lm];
  float* o00 = GX + (size_t)(rowBase + ((lane >> 4) << 3)) * G + colBase + lm;
  float* o10 = o00 + (size_t)16 * G;
#pragma unroll
  for (int i = 0; i < 8; ++i) {
    o00[(size_t)i * G]      = acc00[i] + bv0;
    o00[(size_t)i * G + 16] = acc01[i] + bv1;
    o10[(size_t)i * G]      = acc10[i] + bv0;
    o10[(size_t)i * G + 16] = acc11[i] + bv1;
  }
}

// ------------- one recurrent step: LDS-staged GEMM (h @ Whh^T) + LSTM cell -------
// 64 WGs (one per 16 hidden cols), 16 waves = 4 rowTiles x 4 gates.
// Full h (128KB) + 64 Whh rows (128KB) staged into LDS via async-to-LDS (ASYNCcnt),
// then 4x-reused from ds_load_b128 by the WMMA loop.
__global__ __launch_bounds__(512, 1) void lstm_step_kernel(const __bf16* __restrict__ Whh,
                                                           const float* __restrict__ GX,
                                                           const int* __restrict__ lengths,
                                                           const __bf16* __restrict__ h_in,
                                                           __bf16* __restrict__ h_out,
                                                           float* __restrict__ c,
                                                           int t) {
  extern __shared__ char smem[];
  __bf16* hb = (__bf16*)smem;                         // [64][1024] staged h
  __bf16* wb = (__bf16*)(smem + HBUF_BYTES);          // [64][1024] rows = gate*16+r
  float*  gb = (float*)(smem + HBUF_BYTES + WBUF_BYTES); // [64][GBUF_STRIDE]

  const int tid  = threadIdx.x;
  const int wave = tid >> 5, lane = tid & 31;
  const int gate = wave >> 2, rowTile = wave & 3;
  const int n0 = blockIdx.x << 4;       // hidden column base
  const float* gx = GX + (size_t)t * Bq * G;

  // pull this step's gates_x slice toward the WGP while we compute
  if (tid < 256)
    __builtin_prefetch(gx + (size_t)(tid & 63) * G + ((tid >> 6) << 10) + n0, 0, 1);

  // stage h (contiguous 128KB) and the 64 needed Whh rows into LDS, async
  {
    const unsigned hb_lds = (unsigned)(uintptr_t)hb;   // low 32 bits = LDS offset
    const unsigned wb_lds = (unsigned)(uintptr_t)wb;
    const v8bf* hsrc = (const v8bf*)h_in;
#pragma unroll
    for (int i = tid; i < 8192; i += 512)              // 8192 x 16B chunks
      async_copy16(hb_lds + (unsigned)i * 16u, hsrc + i);
#pragma unroll
    for (int i = tid; i < 8192; i += 512) {
      const int lrow = i >> 7, cc = (i & 127) << 3;    // 128 x 16B chunks per row
      const int grow = ((lrow >> 4) << 10) + n0 + (lrow & 15);
      async_copy16(wb_lds + (unsigned)i * 16u, Whh + (size_t)grow * Hq + cc);
    }
  }
  wait_async0();
  __syncthreads();

  const int rowBase = rowTile << 4;
  const int lm = lane & 15, kh = (lane >> 4) << 3;
  const __bf16* ap = hb + (rowBase + lm) * Hq + kh;
  const __bf16* bp = wb + ((gate << 4) + lm) * Hq + kh;
  v8f acc = {};
#pragma unroll 4
  for (int k = 0; k < Hq; k += 32) {
    v16bf a = cat8(*(const v8bf*)(ap + k), *(const v8bf*)(ap + k + 16));
    v16bf b = cat8(*(const v8bf*)(bp + k), *(const v8bf*)(bp + k + 16));
    acc = __builtin_amdgcn_wmma_f32_16x16x32_bf16(false, a, false, b, (short)0, acc,
                                                  false, false);
  }
  const int crow = rowBase + ((lane >> 4) << 3);
  const int ccol = (gate << 4) + lm;
#pragma unroll
  for (int i = 0; i < 8; ++i) gb[(crow + i) * GBUF_STRIDE + ccol] = acc[i];
  __syncthreads();

#pragma unroll
  for (int e = tid; e < Bq * 16; e += 512) {
    const int b = e >> 4, nl = e & 15;
    const int n = n0 + nl;
    const size_t gr = (size_t)b * G + n;
    float ig = gb[b * GBUF_STRIDE + nl]      + gx[gr];
    float fg = gb[b * GBUF_STRIDE + 16 + nl] + gx[gr + (size_t)Hq];
    float gg = gb[b * GBUF_STRIDE + 32 + nl] + gx[gr + (size_t)2 * Hq];
    float og = gb[b * GBUF_STRIDE + 48 + nl] + gx[gr + (size_t)3 * Hq];
    float iv = sigm(ig), fv = sigm(fg), gv = tanhf(gg), ov = sigm(og);
    const size_t hn = (size_t)b * Hq + n;
    float cOld = c[hn];
    float cNew = fv * cOld + iv * gv;
    float hNew = ov * tanhf(cNew);
    bool m = t < lengths[b];
    c[hn] = m ? cNew : cOld;
    h_out[hn] = (__bf16)(m ? hNew : (float)hb[hn]);   // h_old comes from LDS
  }
}

// ------------- epilogue: bf16 h -> fp32 out -------------
__global__ void final_kernel(const __bf16* __restrict__ h, float* __restrict__ out) {
  int i = blockIdx.x * blockDim.x + threadIdx.x;
  out[i] = (float)h[i];
}

extern "C" void kernel_launch(void* const* d_in, const int* in_sizes, int n_in,
                              void* d_out, int out_size, void* d_ws, size_t ws_size,
                              hipStream_t stream) {
  (void)in_sizes; (void)n_in; (void)out_size; (void)ws_size;
  const int*   tokens  = (const int*)  d_in[0];
  const int*   lengths = (const int*)  d_in[1];
  const float* emb     = (const float*)d_in[2];
  const float* WihF    = (const float*)d_in[3];
  const float* WhhF    = (const float*)d_in[4];
  const float* bih     = (const float*)d_in[5];
  const float* bhh     = (const float*)d_in[6];
  float* out = (float*)d_out;

  char* ws = (char*)d_ws;
  size_t off = 0;
  float*  GX   = (float*) (ws + off); off += (size_t)Tq * Bq * G * sizeof(float);   // 256 MB
  __bf16* Whh  = (__bf16*)(ws + off); off += (size_t)G * Hq * sizeof(__bf16);       // 8 MB
  __bf16* Wih  = (__bf16*)(ws + off); off += (size_t)G * Dq * sizeof(__bf16);       // 4 MB
  __bf16* X    = (__bf16*)(ws + off); off += (size_t)Tq * Bq * Dq * sizeof(__bf16); // 16 MB
  float*  bias = (float*) (ws + off); off += (size_t)G * sizeof(float);
  __bf16* h0   = (__bf16*)(ws + off); off += (size_t)Bq * Hq * sizeof(__bf16);
  __bf16* h1   = (__bf16*)(ws + off); off += (size_t)Bq * Hq * sizeof(__bf16);
  float*  cbuf = (float*) (ws + off); off += (size_t)Bq * Hq * sizeof(float);
  __bf16* hbuf[2] = {h0, h1};

  // 1) weight conversion + bias + state init  (covers G*Hq = 4M elements)
  prep_kernel<<<dim3((G * Hq) / 256), dim3(256), 0, stream>>>(
      WihF, WhhF, bih, bhh, Wih, Whh, bias, h0, cbuf);

  // 2) embedding gather -> bf16  (T*B*D/8 chunks)
  embed_kernel<<<dim3((Tq * Bq * Dq / 8) / 256), dim3(256), 0, stream>>>(tokens, emb, X);

  // 3) input projection GEMM: 512x128 wave-blocks of 32x32, 8 waves/WG
  gemmx_kernel<<<dim3((512 * 128) / 8), dim3(256), 0, stream>>>(X, Wih, bias, GX);

  // 4) 256 sequential recurrent steps (double-buffered h), async-LDS-staged operands
  for (int t = 0; t < Tq; ++t) {
    lstm_step_kernel<<<dim3(Hq / 16), dim3(512), STEP_LDS, stream>>>(
        Whh, GX, lengths, hbuf[t & 1], hbuf[(t + 1) & 1], cbuf, t);
  }

  // 5) final h (lands in buffer 0 after 256 steps) -> fp32 out
  final_kernel<<<dim3((Bq * Hq) / 256), dim3(256), 0, stream>>>(hbuf[0], out);
}
